// StyleLayer_51573967290887
// MI455X (gfx1250) — compile-verified, matrix operand order checked by reading, not run on hardware
//
#include <hip/hip_runtime.h>

typedef __bf16 bf16;
typedef __attribute__((ext_vector_type(8)))  __bf16 v8bf;
typedef __attribute__((ext_vector_type(16))) __bf16 v16bf;
typedef __attribute__((ext_vector_type(8)))  float  v8f;

typedef int v4i __attribute__((vector_size(16)));
typedef __attribute__((address_space(1))) v4i gv4i;
typedef __attribute__((address_space(3))) v4i lv4i;

#if __has_builtin(__builtin_amdgcn_global_load_async_to_lds_b128) && \
    __has_builtin(__builtin_amdgcn_s_wait_asynccnt)
#define USE_ASYNC 1
#else
#define USE_ASYNC 0
#endif

#define B_     8
#define CIN_   512
#define COUT_  512
#define SDIM_  512
#define RES_   64
#define HW_    (RES_*RES_)

#define AFF_SCALE  0.04419417382415922f    /* 1/sqrt(512) */
#define CONV_SCALE 0.014731391274719742f   /* 1/sqrt(512*9) */
#define LRELU_GAIN 1.41421356237309515f

// LDS row stride in bf16 elements: 40 (=80 bytes, 16B aligned, conflict-free)
#define LROW 40

// ---------------- small precompute kernels ----------------

__global__ __launch_bounds__(256) void style_kernel(
    const float* __restrict__ w, const float* __restrict__ aw,
    const float* __restrict__ ab, float* __restrict__ style) {
  int idx = blockIdx.x * 256 + threadIdx.x;       // B*CIN = 4096
  int b = idx >> 9, c = idx & 511;
  float s = 0.f;
  for (int k = 0; k < SDIM_; ++k) s += w[b * SDIM_ + k] * aw[c * SDIM_ + k];
  style[idx] = s * AFF_SCALE + ab[c];
}

__global__ __launch_bounds__(256) void wsq_kernel(
    const float* __restrict__ cw, float* __restrict__ wsq) {
  int idx = blockIdx.x * 256 + threadIdx.x;       // COUT*CIN = 262144
  const float* p = cw + (size_t)idx * 9;
  float s = 0.f;
#pragma unroll
  for (int t = 0; t < 9; ++t) s += p[t] * p[t];
  wsq[idx] = s;
}

__global__ __launch_bounds__(256) void demod_kernel(
    const float* __restrict__ wsq, const float* __restrict__ style,
    float* __restrict__ dmod) {
  int idx = blockIdx.x * 256 + threadIdx.x;       // B*COUT = 4096
  int b = idx >> 9, o = idx & 511;
  float s = 0.f;
  for (int i = 0; i < CIN_; ++i) {
    float st = style[b * CIN_ + i];
    s += wsq[o * CIN_ + i] * st * st;
  }
  dmod[idx] = rsqrtf(s * CONV_SCALE * CONV_SCALE + 1e-8f);
}

// x_mod[b][cin][h][w] = bf16( x * style[b][cin] * CONV_SCALE )
__global__ __launch_bounds__(256) void modx_kernel(
    const float* __restrict__ x, const float* __restrict__ style,
    bf16* __restrict__ xb) {
  int idx = blockIdx.x * 256 + threadIdx.x;       // B*CIN*HW = 16777216
  int b = idx >> 21;
  int cin = (idx >> 12) & 511;
  xb[idx] = (bf16)(x[idx] * style[b * CIN_ + cin] * CONV_SCALE);
}

// wbt[tap][cout][cin] = bf16( conv_w[cout][cin][tap] ), tap = ky*3+kx
__global__ __launch_bounds__(256) void wre_kernel(
    const float* __restrict__ cw, bf16* __restrict__ wbt) {
  int idx = blockIdx.x * 256 + threadIdx.x;       // 9*COUT*CIN = 2359296
  int tap = idx >> 18;
  int rem = idx & 262143;
  int o = rem >> 9, i = rem & 511;
  wbt[idx] = (bf16)cw[((size_t)(o * CIN_ + i)) * 9 + tap];
}

// ---------------- main implicit-GEMM conv kernel ----------------
// grid (64, 4, 8) = (h, coutTile128, b); block 256 = 8 wave32
// block output: 128 couts x 64 pixels (one image row)
// wave w: couts [w*16, w*16+16) x all 64 px (4 WMMA n-tiles; A reused 4x)

__global__ __launch_bounds__(256) void conv_kernel(
    const bf16*  __restrict__ xb,     // [B][CIN][64][64] modulated bf16
    const bf16*  __restrict__ wbt,    // [9][COUT][CIN]   bf16
    const float* __restrict__ dmod,   // [B][COUT]
    const float* __restrict__ noise,  // [B][64][64]
    const float* __restrict__ nscale, // scalar
    const float* __restrict__ bias,   // [COUT]
    float*       __restrict__ out) {  // [B][COUT][64][64]
  const int h     = blockIdx.x;
  const int cout0 = blockIdx.y * 128;
  const int b     = blockIdx.z;

  __shared__ __align__(16) bf16 Wlds[3][128 * LROW]; // 3 taps, 128 cout x 32 cin
  __shared__ __align__(16) bf16 Xlds[66 * LROW];     // 66 halo px x 32 cin (transposed)
  __shared__ float Dlds[128];

  const int tid  = threadIdx.x;
  const int lane = tid & 31;
  const int wave = tid >> 5;           // 0..7 : 16-cout group
  const int lrow = lane & 15;
  const int kb   = (lane < 16) ? 0 : 8;
  const int hi16 = (lane >= 16) ? 8 : 0;

  if (tid < LROW) {                    // zero halo rows (px=-1, px=64), set once
    Xlds[0 * LROW + tid]  = (bf16)0.0f;
    Xlds[65 * LROW + tid] = (bf16)0.0f;
  }
  if (tid < 128) Dlds[tid] = dmod[b * COUT_ + cout0 + tid];

  v8f acc[4];
#pragma unroll
  for (int i = 0; i < 4; ++i) acc[i] = v8f{};

  const int xc = tid >> 3;             // 0..31  cin-in-chunk for X loads
  const int xs = (tid & 7) * 8;        // 0..56  px segment start

  for (int cc = 0; cc < CIN_; cc += 32) {
    for (int dy = 0; dy < 3; ++dy) {
      __syncthreads();
      // ---- stage X: one row (h+dy-1) of 32 cins, transposed into [px][cin]
      const int hin = h + dy - 1;
      v8bf xv = {};
      if (hin >= 0 && hin < RES_) {
        xv = *reinterpret_cast<const v8bf*>(
            xb + ((((size_t)b * CIN_ + cc + xc) * RES_ + hin) * RES_ + xs));
      }
#pragma unroll
      for (int i = 0; i < 8; ++i) Xlds[(xs + 1 + i) * LROW + xc] = xv[i];
      // ---- stage W: 3 taps of this dy, 128 cout x 32 cin each
      //      1536 16B segments / 256 threads = 6 per thread
#pragma unroll
      for (int j = 0; j < 6; ++j) {
        const int task = tid + 256 * j;
        const int tt   = task >> 9;           // 0..2
        const int rem  = task & 511;
        const int row  = rem >> 2;            // 0..127
        const int seg  = (rem & 3) * 8;       // cin sub-offset
        const bf16* gp =
            wbt + (((size_t)(dy * 3 + tt) * COUT_ + cout0 + row) * CIN_ + cc + seg);
        bf16* lp = &Wlds[tt][row * LROW + seg];
#if USE_ASYNC
        __builtin_amdgcn_global_load_async_to_lds_b128((gv4i*)gp, (lv4i*)lp, 0,
                                                       0);
#else
        *reinterpret_cast<v8bf*>(lp) = *reinterpret_cast<const v8bf*>(gp);
#endif
      }
#if USE_ASYNC
      __builtin_amdgcn_s_wait_asynccnt(0);
#endif
      __syncthreads();
      // ---- compute: 3 dx taps; A loaded once, reused over 4 n-tiles
#pragma unroll
      for (int dx = 0; dx < 3; ++dx) {
        const bf16* ap = &Wlds[dx][(wave * 16 + lrow) * LROW + kb];
        v8bf alo = *reinterpret_cast<const v8bf*>(ap);
        v8bf ahi = *reinterpret_cast<const v8bf*>(ap + 16);
        v16bf a = __builtin_shufflevector(alo, ahi, 0, 1, 2, 3, 4, 5, 6, 7, 8,
                                          9, 10, 11, 12, 13, 14, 15);
#pragma unroll
        for (int nt = 0; nt < 4; ++nt) {
          const bf16* bp = &Xlds[(nt * 16 + lrow + dx) * LROW + kb];
          v8bf blo = *reinterpret_cast<const v8bf*>(bp);
          v8bf bhi = *reinterpret_cast<const v8bf*>(bp + 16);
          v16bf bv = __builtin_shufflevector(blo, bhi, 0, 1, 2, 3, 4, 5, 6, 7,
                                             8, 9, 10, 11, 12, 13, 14, 15);
          acc[nt] = __builtin_amdgcn_wmma_f32_16x16x32_bf16(
              false, a, false, bv, (short)0, acc[nt], false, false);
        }
      }
    }
  }

  // ---- epilogue: demod scale, noise, bias, lrelu * sqrt(2)
  const float ns = nscale[0];
#pragma unroll
  for (int nt = 0; nt < 4; ++nt) {
    const int wpx = nt * 16 + lrow;
    const float nz = noise[(size_t)b * HW_ + h * RES_ + wpx] * ns;
#pragma unroll
    for (int r = 0; r < 8; ++r) {
      const int m = wave * 16 + hi16 + r;
      float v = acc[nt][r] * Dlds[m] + nz + bias[cout0 + m];
      v = (v > 0.0f ? v : 0.2f * v) * LRELU_GAIN;
      out[(((size_t)b * COUT_ + cout0 + m) * RES_ + h) * RES_ + wpx] = v;
    }
  }
}

// ---------------- launch ----------------

extern "C" void kernel_launch(void* const* d_in, const int* in_sizes, int n_in,
                              void* d_out, int out_size, void* d_ws,
                              size_t ws_size, hipStream_t stream) {
  const float* x      = (const float*)d_in[0];
  const float* w      = (const float*)d_in[1];
  const float* noise  = (const float*)d_in[2];
  const float* aw     = (const float*)d_in[3];
  const float* ab     = (const float*)d_in[4];
  const float* cw     = (const float*)d_in[5];
  const float* nscale = (const float*)d_in[6];
  const float* bias   = (const float*)d_in[7];
  float* out = (float*)d_out;

  char* wsb = (char*)d_ws;
  float* style = (float*)(wsb + 0);               // 16 KB
  float* dmod  = (float*)(wsb + 16384);           // 16 KB
  float* wsq   = (float*)(wsb + 32768);           // 1 MB
  bf16*  wbt   = (bf16*)(wsb + 1081344);          // 4.5 MB  [9][512][512]
  bf16*  xb    = (bf16*)(wsb + 5799936);          // 32 MB   [8][512][64][64]

  style_kernel<<<16, 256, 0, stream>>>(w, aw, ab, style);
  wsq_kernel<<<1024, 256, 0, stream>>>(cw, wsq);
  demod_kernel<<<16, 256, 0, stream>>>(wsq, style, dmod);
  modx_kernel<<<65536, 256, 0, stream>>>(x, style, xb);
  wre_kernel<<<9216, 256, 0, stream>>>(cw, wbt);

  conv_kernel<<<dim3(64, 4, 8), 256, 0, stream>>>(xb, wbt, dmod, noise, nscale,
                                                  bias, out);
}
